// Int8Linear_70239895159253
// MI455X (gfx1250) — matrix operand order repaired and checked by reading, not compiled, not run on hardware
//
#include <hip/hip_runtime.h>
#include <stdint.h>

typedef _Float16 v16h __attribute__((ext_vector_type(16)));
typedef _Float16 v8h  __attribute__((ext_vector_type(8)));
typedef float    v8f  __attribute__((ext_vector_type(8)));
typedef int      iv4  __attribute__((ext_vector_type(4)));
typedef unsigned uv4  __attribute__((ext_vector_type(4)));
typedef int      iv8  __attribute__((ext_vector_type(8)));

// ---- gfx1250 async global->LDS path (probe-verified on this toolchain) ----
#if defined(__gfx1250__) && __has_builtin(__builtin_amdgcn_global_load_async_to_lds_b128)
#define ASYNC_LDS 1
#else
#define ASYNC_LDS 0
#endif

// ---- gfx1250 Tensor Data Mover path (guarded; falls back to async) ----
#if defined(__gfx1250__) && __has_builtin(__builtin_amdgcn_tensor_load_to_lds)
#define TDM_LDS 1
#else
#define TDM_LDS 0
#endif

#if ASYNC_LDS
__device__ inline void async_copy_b128(const void* g, void* l) {
  auto gp = reinterpret_cast<__attribute__((address_space(1))) iv4*>(
      reinterpret_cast<uintptr_t>(g));
  auto lp = reinterpret_cast<__attribute__((address_space(3))) iv4*>(
      (uint32_t)reinterpret_cast<uintptr_t>(l));
  __builtin_amdgcn_global_load_async_to_lds_b128(gp, lp, 0, 0);
}
#if __has_builtin(__builtin_amdgcn_s_wait_asynccnt)
#define WAIT_ASYNC() __builtin_amdgcn_s_wait_asynccnt(0)
#else
#define WAIT_ASYNC() asm volatile("s_wait_asynccnt 0" ::: "memory")
#endif
#endif

#if TDM_LDS
#if __has_builtin(__builtin_amdgcn_s_wait_tensorcnt)
#define WAIT_TENSOR() __builtin_amdgcn_s_wait_tensorcnt(0)
#else
#define WAIT_TENSOR() asm volatile("s_wait_tensorcnt 0" ::: "memory")
#endif
// DMA one 128-row x 64-half tile (row stride = strideElems halves) into LDS,
// with 8-half row padding (pad_interval=32 DWORDs, pad_amount=4 DWORDs -> LDS
// row pitch 72 halves). D# packing per CDNA5 ISA §8.3/8.4.
__device__ inline void tdm_load_tile(const void* gaddr, void* lds, int strideElems) {
  const uint64_t ga = (uint64_t)(uintptr_t)gaddr;
  const uint32_t la = (uint32_t)(uintptr_t)lds;
  uv4 g0;
  g0[0] = 1u;                                                  // count=1 (valid user D#)
  g0[1] = la;                                                  // lds_addr (bytes)
  g0[2] = (uint32_t)ga;                                        // global_addr[31:0]
  g0[3] = (uint32_t)((ga >> 32) & 0x01FFFFFFu) | (2u << 30);   // addr[56:32] | type=2
  iv8 g1;
  g1[0] = (1 << 16)      // data_size = 1 -> 2 bytes
        | (1 << 20)      // pad_enable
        | (4 << 22)      // pad_interval: 32 DWORDs (=128B row)
        | (3 << 25);     // pad_amount:  4 DWORDs (=8 halves)
  g1[1] = (int)(64u << 16);    // tensor_dim0 = 64  (bits 79:48)
  g1[2] = (int)(128u << 16);   // tensor_dim1 = 128 (bits 111:80)
  g1[3] = (int)(64u << 16);    // tile_dim0 = 64    (bits 127:112)
  g1[4] = 128;                 // tile_dim1 = 128, tile_dim2 = 0
  g1[5] = strideElems;         // tensor_dim0_stride[31:0]
  g1[6] = 0;
  g1[7] = 0;
  iv4 z4 = {0, 0, 0, 0};
#if defined(__clang_major__) && __clang_major__ >= 23
  iv8 z8 = {0, 0, 0, 0, 0, 0, 0, 0};
  __builtin_amdgcn_tensor_load_to_lds(g0, g1, z4, z4, z8, 0);
#else
  __builtin_amdgcn_tensor_load_to_lds(g0, g1, z4, z4, 0);
#endif
}
#endif

namespace {
constexpr int BM = 128, BN = 128, BK = 64;
constexpr int LDA = BK + 8;       // 72 halves; pad kills LDS bank conflicts
constexpr int LDB = BK + 8;       // sB is [n][k]
constexpr int LDC = BN + 8;       // 136 halves, epilogue staging stride
constexpr int THREADS = 256;      // 8 wave32
constexpr int WAVES_M = 2, WAVES_N = 4;
constexpr int WM = BM / WAVES_M;  // 64
constexpr int WN = BN / WAVES_N;  // 32
constexpr int MT = WM / 16;       // 4
constexpr int NT = WN / 16;       // 2
}

// One-time weight dequant + transpose: W[K][N] int32 -> Wt[N][K] fp16 (exact).
__global__ __launch_bounds__(256)
void dequant_transpose(const int* __restrict__ W, _Float16* __restrict__ Wt,
                       int K, int N) {
  __shared__ _Float16 t[32][33];
  const int n0 = blockIdx.x * 32, k0 = blockIdx.y * 32;
  const int tx = threadIdx.x & 31, ty = threadIdx.x >> 5;  // 32 x 8
#pragma unroll
  for (int i = 0; i < 32; i += 8)
    t[ty + i][tx] = (_Float16)W[(long)(k0 + ty + i) * N + (n0 + tx)];
  __syncthreads();
#pragma unroll
  for (int i = 0; i < 32; i += 8)
    Wt[(long)(n0 + ty + i) * K + (k0 + tx)] = t[tx][ty + i];
}

template <bool USE_WT>
__global__ __launch_bounds__(THREADS)
void int8_linear_wmma(const _Float16* __restrict__ X,    // [M,K] fp16
                      const int* __restrict__ W,         // [K,N] int32 (if !USE_WT)
                      const _Float16* __restrict__ Wt,   // [N,K] fp16 (if USE_WT)
                      const _Float16* __restrict__ Bv,   // [N] fp16
                      _Float16* __restrict__ O,          // [M,N] fp16
                      int M, int N, int K) {
  __shared__ alignas(16) _Float16 sA[2][BM * LDA];  // [m][k]
  __shared__ alignas(16) _Float16 sB[2][BN * LDB];  // [n][k]

  const int tid = threadIdx.x, lane = tid & 31, wave = tid >> 5;
  const int waveM = wave >> 2, waveN = wave & 3;
  const int m0 = blockIdx.y * BM, n0 = blockIdx.x * BN;

  const int aRow = tid >> 3;          // 0..31 (+32 per pass, 4 passes); also Wt n-row
  const int aCol = (tid & 7) * 8;     // 0..56 halves
  const int bRow = tid >> 5;          // int32 path: k-row (+8 per pass, 8 passes)
  const int bCol = (tid & 31) * 4;    // int32 path: n-col
  const int nk = K / BK;              // 64

  v8f acc[MT][NT] = {};
  v8h aReg[4], wReg[4];
  iv4 bReg[8];

  // Persistent per-thread global pointers, advanced by BK each tile.
  const _Float16* gA = X + (long)(m0 + aRow) * K + aCol;
  const _Float16* gBw = USE_WT ? (Wt + (long)(n0 + aRow) * K + aCol) : nullptr;
  const int* gW = USE_WT ? nullptr : (W + (long)bRow * N + n0 + bCol);
  // Block-uniform tile base pointers (TDM descriptors).
  const _Float16* gAblk = X + (long)m0 * K;
  const _Float16* gBblk = USE_WT ? (Wt + (long)n0 * K) : nullptr;
  const long rowStepA = 32L * K;   // A / Wt pass stride
  const long rowStepW = 8L * N;    // int32 W pass stride

  auto issueTile = [&](int buf) {
#if TDM_LDS
    if constexpr (USE_WT) {
      if (wave == 0) {
        tdm_load_tile(gAblk, &sA[buf][0], K);
        tdm_load_tile(gBblk, &sB[buf][0], K);
      }
      return;
    }
#endif
#if ASYNC_LDS
#pragma unroll
    for (int p = 0; p < 4; ++p)
      async_copy_b128(gA + p * rowStepA, &sA[buf][(aRow + 32 * p) * LDA + aCol]);
#else
#pragma unroll
    for (int p = 0; p < 4; ++p) aReg[p] = *(const v8h*)(gA + p * rowStepA);
#endif
    if constexpr (USE_WT) {
#if ASYNC_LDS
#pragma unroll
      for (int p = 0; p < 4; ++p)
        async_copy_b128(gBw + p * rowStepA, &sB[buf][(aRow + 32 * p) * LDB + aCol]);
#else
#pragma unroll
      for (int p = 0; p < 4; ++p) wReg[p] = *(const v8h*)(gBw + p * rowStepA);
#endif
    } else {
#pragma unroll
      for (int p = 0; p < 8; ++p) bReg[p] = *(const iv4*)(gW + p * rowStepW);
    }
  };

  auto advance = [&] {
    gA += BK;
    gAblk += BK;
    if constexpr (USE_WT) { gBw += BK; gBblk += BK; }
    else gW += (long)BK * N;
  };

  auto commitTile = [&](int buf) {
#if !ASYNC_LDS && !TDM_LDS
#pragma unroll
    for (int p = 0; p < 4; ++p)
      *(v8h*)(&sA[buf][(aRow + 32 * p) * LDA + aCol]) = aReg[p];
#endif
    if constexpr (USE_WT) {
#if !ASYNC_LDS && !TDM_LDS
#pragma unroll
      for (int p = 0; p < 4; ++p)
        *(v8h*)(&sB[buf][(aRow + 32 * p) * LDB + aCol]) = wReg[p];
#endif
    } else {
#pragma unroll
      for (int p = 0; p < 8; ++p) {
        const int kr = bRow + p * 8;
#pragma unroll
        for (int j = 0; j < 4; ++j)  // dequant s32 -> f16 (exact in int8 range)
          sB[buf][(bCol + j) * LDB + kr] = (_Float16)bReg[p][j];
      }
    }
  };

  auto tileFence = [&] {
#if TDM_LDS
    if constexpr (USE_WT) {
      if (wave == 0) WAIT_TENSOR();
      __syncthreads();
      return;
    }
#endif
#if ASYNC_LDS
    WAIT_ASYNC();
#endif
    __syncthreads();
  };

  // ---- prologue ----
  issueTile(0);
  advance();
  commitTile(0);
  tileFence();

  const int rsel  = lane & 15;
  const int kselA = (lane >> 4) * 8;   // A: lanes 0-15 K 0..7/16..23, 16-31 K 8..15/24..31
  const int kselB = (lane >> 4) * 8;   // B: lanes 0-15 K 0..7/16..23, 16-31 K 8..15/24..31

  for (int kt = 0; kt < nk; ++kt) {
    const int cur = kt & 1;
    const bool more = (kt + 1 < nk);

    if (more) { issueTile(1 - cur); advance(); }
    if (kt + 2 < nk) {  // pointers now at tile kt+2: L2 prefetch for free
      __builtin_prefetch(gA, 0, 1);
      if constexpr (USE_WT) __builtin_prefetch(gBw, 0, 1);
      else                  __builtin_prefetch(gW, 0, 1);
    }

    const _Float16* As = sA[cur];
    const _Float16* Bs = sB[cur];

#pragma unroll
    for (int ks = 0; ks < 2; ++ks) {     // two 32-K substeps per tile
      const int kb = ks * 32;
      v16h a[MT], b[NT];
#pragma unroll
      for (int mt = 0; mt < MT; ++mt) {
        const _Float16* p = As + (waveM * WM + mt * 16 + rsel) * LDA + kb + kselA;
        v8h lo = *(const v8h*)(p);
        v8h hi = *(const v8h*)(p + 16);
        a[mt] = __builtin_shufflevector(lo, hi, 0,1,2,3,4,5,6,7,8,9,10,11,12,13,14,15);
      }
#pragma unroll
      for (int nt = 0; nt < NT; ++nt) {
        const _Float16* p = Bs + (waveN * WN + nt * 16 + rsel) * LDB + kb + kselB;
        v8h lo = *(const v8h*)(p);
        v8h hi = *(const v8h*)(p + 16);
        b[nt] = __builtin_shufflevector(lo, hi, 0,1,2,3,4,5,6,7,8,9,10,11,12,13,14,15);
      }
#pragma unroll
      for (int mt = 0; mt < MT; ++mt)
#pragma unroll
        for (int nt = 0; nt < NT; ++nt)
          acc[mt][nt] = __builtin_amdgcn_wmma_f32_16x16x32_f16(
              false, a[mt], false, b[nt], (short)0, acc[mt][nt], false, false);
    }

    if (more) commitTile(1 - cur);
    tileFence();
  }

  // ---- epilogue: fuse bias, stage C tile in (dead) sA, coalesced b128 stores ----
  _Float16* sC = (_Float16*)sA;  // 128*LDC*2 = 34,816 B <= sizeof(sA)
#pragma unroll
  for (int mt = 0; mt < MT; ++mt) {
    const int r0 = waveM * WM + mt * 16 + (lane >> 4) * 8;
#pragma unroll
    for (int nt = 0; nt < NT; ++nt) {
      const int c = waveN * WN + nt * 16 + (lane & 15);
      const float bv = (float)Bv[n0 + c];
#pragma unroll
      for (int r = 0; r < 8; ++r)
        sC[(r0 + r) * LDC + c] = (_Float16)(acc[mt][nt][r] + bv);
    }
  }
  __syncthreads();
  const int cRow = tid >> 4;         // 0..15 (+16 per pass)
  const int cCol = (tid & 15) * 8;   // halves
#pragma unroll
  for (int p = 0; p < 8; ++p) {
    const int r = cRow + p * 16;
    v8h v = *(const v8h*)(&sC[r * LDC + cCol]);
    *(v8h*)(O + (long)(m0 + r) * N + n0 + cCol) = v;
  }
}

extern "C" void kernel_launch(void* const* d_in, const int* in_sizes, int n_in,
                              void* d_out, int out_size, void* d_ws, size_t ws_size,
                              hipStream_t stream) {
  const _Float16* x    = (const _Float16*)d_in[0];
  const int*      w    = (const int*)d_in[1];
  const _Float16* bias = (const _Float16*)d_in[2];
  _Float16*       out  = (_Float16*)d_out;

  const int N = in_sizes[2];
  const int K = in_sizes[1] / N;
  const int M = in_sizes[0] / K;

  dim3 grid(N / BN, M / BM), block(THREADS);
  const size_t wtBytes = (size_t)N * (size_t)K * sizeof(_Float16);

  if (ws_size >= wtBytes) {
    _Float16* wt = (_Float16*)d_ws;
    dim3 g2(N / 32, K / 32);
    dequant_transpose<<<g2, dim3(256), 0, stream>>>(w, wt, K, N);
    int8_linear_wmma<true><<<grid, block, 0, stream>>>(x, w, wt, bias, out, M, N, K);
  } else {
    int8_linear_wmma<false><<<grid, block, 0, stream>>>(x, w, (const _Float16*)nullptr,
                                                        bias, out, M, N, K);
  }
}
